// CRF1_25675314495495
// MI455X (gfx1250) — matrix operand order repaired and checked by reading, not compiled, not run on hardware
//
#include <hip/hip_runtime.h>
#include <hip/hip_bf16.h>

typedef __attribute__((ext_vector_type(16))) _Float16 v16h;
typedef __attribute__((ext_vector_type(8)))  _Float16 v8h;
typedef __attribute__((ext_vector_type(8)))  float    v8f;

union V16 { v16h v; v8h h[2]; };

#define T_STEPS 1024
#define BATCH   512
#define K_TAG   128

#define LOG2E 1.4426950408889634f
#define LN2   0.6931471805599453f

// Raw hardware transcendentals (operands are guaranteed normal-range here).
// Device pass uses the amdgcn builtins (v_exp_f32 / v_log_f32, no denorm
// guard sequence); host pass only needs to parse these __device__ bodies.
__device__ __forceinline__ float hw_exp2(float x) {
#if defined(__HIP_DEVICE_COMPILE__)
  return __builtin_amdgcn_exp2f(x);
#else
  return x;
#endif
}
__device__ __forceinline__ float hw_log2(float x) {
#if defined(__HIP_DEVICE_COMPILE__)
  return __builtin_amdgcn_logf(x);
#else
  return x;
#endif
}

// ---------------------------------------------------------------------------
// Kernel A: PT[j][i] = (f16) exp(transitions[i][j])   (transposed, row-major j)
// ---------------------------------------------------------------------------
__global__ __launch_bounds__(256) void crf_expT_kernel(
    const float* __restrict__ trans, _Float16* __restrict__ PT) {
  int idx = blockIdx.x * 256 + threadIdx.x;      // 0..16383
  int j = idx >> 7, i = idx & 127;
  PT[idx] = (_Float16)__expf(trans[i * K_TAG + j]);
}

// ---------------------------------------------------------------------------
// Kernel B: forward scan (log-partition) in the BASE-2 domain.
// lp2 = lp * log2(e);  step: lp2' = m2 + log2( 2^(lp2-m2) x exp(trans) ) + em*log2e
// One WG = 16 batch rows, 8 waves; wave owns a 16-column tile; exp(trans)^T
// panel resident in VGPRs. Stabilizer = one-step-stale row max (tracked via
// ds_max_num_f32 atomics into a rotating 3-buffer, off the critical path);
// exponent clamped at 15.5 so f16 never overflows (2^15.5 < 65504).
// ---------------------------------------------------------------------------
__global__ __launch_bounds__(256) void crf_forward_kernel(
    const float* __restrict__ emissions, const float* __restrict__ mask,
    const float* __restrict__ start, const float* __restrict__ endt,
    const _Float16* __restrict__ PT, float* __restrict__ logZ) {
  __shared__ float    lp[16][132];                  // padded (bank conflicts)
  __shared__ __align__(16) _Float16 Apk[16][136];   // row stride 272B, 16B-aligned
  __shared__ float    rm[3][16];                    // rotating row-max buffers
  __shared__ float    mrow[16];

  const int tid   = threadIdx.x;
  const int wave  = tid >> 5;        // 0..7 -> N-tile
  const int lane  = tid & 31;
  const int hi    = lane >> 4;       // 0/1 (lane group)
  const int l16   = lane & 15;
  const int Bbase = blockIdx.x * 16; // 32 blocks * 16 rows = 512 batch
  const int r     = tid >> 4;        // phase-1 row
  const int s     = tid & 15;        // phase-1 column group

  // ---- Resident B panel: columns col = wave*16 + l16 of exp(trans) ----
  // 16-bit B 32x16, contiguous k order used here: lane L holds halves K = (L<16?0:16)+e, e=0..15
  const int col = wave * 16 + l16;
  V16 bch[4];
#pragma unroll
  for (int c = 0; c < 4; ++c) {
    const _Float16* src = PT + (size_t)col * K_TAG + c * 32 + hi * 16;
    bch[c].h[0] = *(const v8h*)(src);
    bch[c].h[1] = *(const v8h*)(src + 8);
  }

  // ---- init: lp2[b][k] = (start[k] + emissions[0][b][k])*log2e; seed rm[1] ----
  if (tid < 16) { rm[0][tid] = -3.0e38f; rm[1][tid] = -3.0e38f; rm[2][tid] = -3.0e38f; }
  __syncthreads();
  {
    const float* e0 = emissions + ((size_t)(Bbase + r)) * K_TAG;
    float mx = -3.0e38f;
#pragma unroll
    for (int i = 0; i < 8; ++i) {
      int k = s * 8 + i;
      float v = (start[k] + e0[k]) * LOG2E;
      lp[r][k] = v;
      mx = fmaxf(mx, v);
    }
    __hip_atomic_fetch_max(&rm[1][r], mx, __ATOMIC_RELAXED, __HIP_MEMORY_SCOPE_WORKGROUP);
  }
  __syncthreads();

  int ir = 1;                        // rm[ir]: stale max (read)
  for (int t = 1; t < T_STEPS; ++t) {
    const int iw = (ir + 1 == 3) ? 0 : ir + 1;     // atomic-write buffer
    const int iz = (iw + 1 == 3) ? 0 : iw + 1;     // reset buffer

    // ---- phase 1: pack 2^(lp2 - stale_max) to f16; track true max lazily ----
    {
      float m_st = rm[ir][r];
      float v[8];
      float mx = -3.0e38f;
#pragma unroll
      for (int i = 0; i < 8; ++i) { v[i] = lp[r][s * 8 + i]; mx = fmaxf(mx, v[i]); }
      __hip_atomic_fetch_max(&rm[iw][r], mx, __ATOMIC_RELAXED, __HIP_MEMORY_SCOPE_WORKGROUP);
      if (tid < 16) {
        rm[iz][tid] = -3.0e38f;
        mrow[tid]   = mask[(size_t)t * BATCH + Bbase + tid];
      }
#pragma unroll
      for (int i = 0; i < 8; ++i)
        Apk[r][s * 8 + i] = (_Float16)hw_exp2(fminf(v[i] - m_st, 15.5f));
    }
    __syncthreads();

    // ---- phase 2: emissions early, A loads batched, dual-accumulator WMMA ----
    float emv[8];
    {
      const float* emt = emissions + ((size_t)t * BATCH + Bbase) * K_TAG;
#pragma unroll
      for (int q = 0; q < 8; ++q)
        emv[q] = emt[(size_t)(q + hi * 8) * K_TAG + col];
    }
    V16 a[4];
    {
      const _Float16* arow = &Apk[l16][0];
#pragma unroll
      for (int c = 0; c < 4; ++c) {
        // A 16x32 layout (matches B): elems 0-15 -> K=32c+hi*16+e
        a[c].h[0] = *(const v8h*)(arow + c * 32 + hi * 16);
        a[c].h[1] = *(const v8h*)(arow + c * 32 + hi * 16 + 8);
      }
    }
    v8f acc0 = {}, acc1 = {};
    acc0 = __builtin_amdgcn_wmma_f32_16x16x32_f16(false, a[0].v, false, bch[0].v, (short)0, acc0, false, false);
    acc1 = __builtin_amdgcn_wmma_f32_16x16x32_f16(false, a[1].v, false, bch[1].v, (short)0, acc1, false, false);
    acc0 = __builtin_amdgcn_wmma_f32_16x16x32_f16(false, a[2].v, false, bch[2].v, (short)0, acc0, false, false);
    acc1 = __builtin_amdgcn_wmma_f32_16x16x32_f16(false, a[3].v, false, bch[3].v, (short)0, acc1, false, false);
    v8f acc = acc0 + acc1;

    // prefetch next step's emissions tile (speculative; dropped if OOB)
    if (t + 1 < T_STEPS) {
      __builtin_prefetch(emissions + ((size_t)(t + 1) * BATCH + Bbase + hi * 8) * K_TAG + col, 0, 2);
    }

    // ---- phase 3: lp2' = mk*(m_st + log2(acc) + em*log2e) + (1-mk)*lp2 ----
#pragma unroll
    for (int q = 0; q < 8; ++q) {
      int row = q + hi * 8;                 // C layout: VGPR q, lane group hi
      float mk  = mrow[row];
      float sc  = fmaf(emv[q], LOG2E, rm[ir][row] + hw_log2(acc[q]));
      float old = lp[row][col];
      lp[row][col] = fmaf(sc - old, mk, old);
    }
    __syncthreads();

    ir = iw;
  }

  // ---- final: logZ[b] = ln2 * log2sumexp2_k(lp2[b][k] + end[k]*log2e) ----
  {
    float v[8];
    float mx = -3.0e38f;
#pragma unroll
    for (int i = 0; i < 8; ++i) {
      int k = s * 8 + i;
      v[i] = lp[r][k] + endt[k] * LOG2E;
      mx = fmaxf(mx, v[i]);
    }
#pragma unroll
    for (int off = 8; off >= 1; off >>= 1) mx = fmaxf(mx, __shfl_xor(mx, off, 32));
    float sum = 0.0f;
#pragma unroll
    for (int i = 0; i < 8; ++i) sum += hw_exp2(v[i] - mx);
#pragma unroll
    for (int off = 8; off >= 1; off >>= 1) sum += __shfl_xor(sum, off, 32);
    if (s == 0) logZ[Bbase + r] = LN2 * (mx + hw_log2(sum));
  }
}

// ---------------------------------------------------------------------------
// Kernel C: gold-path score per batch element
// ---------------------------------------------------------------------------
__global__ __launch_bounds__(256) void crf_path_kernel(
    const float* __restrict__ emissions, const int* __restrict__ tags,
    const float* __restrict__ mask, const float* __restrict__ start,
    const float* __restrict__ trans, const float* __restrict__ endt,
    float* __restrict__ path) {
  int b = blockIdx.x * 256 + threadIdx.x;
  if (b >= BATCH) return;
  int prev = tags[b];
  float sc = start[prev];
  int cnt = 0;
  for (int t = 0; t < T_STEPS; ++t) {
    int   tg = tags[(size_t)t * BATCH + b];
    float mk = mask[(size_t)t * BATCH + b];
    sc += emissions[((size_t)t * BATCH + b) * K_TAG + tg] * mk;
    if (t > 0) sc += trans[prev * K_TAG + tg] * mk;
    cnt += (int)mk;
    prev = tg;
  }
  int last = tags[(size_t)(cnt - 1) * BATCH + b];
  sc += endt[last];
  path[b] = sc;
}

// ---------------------------------------------------------------------------
// Kernel D: out = sum_b (path[b] - logZ[b])
// ---------------------------------------------------------------------------
__global__ __launch_bounds__(256) void crf_reduce_kernel(
    const float* __restrict__ path, const float* __restrict__ logZ,
    float* __restrict__ out) {
  __shared__ float red[256];
  int tid = threadIdx.x;
  float s = (path[tid] - logZ[tid]) + (path[tid + 256] - logZ[tid + 256]);
  red[tid] = s;
  __syncthreads();
  for (int off = 128; off > 0; off >>= 1) {
    if (tid < off) red[tid] += red[tid + off];
    __syncthreads();
  }
  if (tid == 0) out[0] = red[0];
}

// ---------------------------------------------------------------------------
extern "C" void kernel_launch(void* const* d_in, const int* in_sizes, int n_in,
                              void* d_out, int out_size, void* d_ws, size_t ws_size,
                              hipStream_t stream) {
  const float* emissions = (const float*)d_in[0];
  const int*   tags      = (const int*)d_in[1];
  const float* mask      = (const float*)d_in[2];
  const float* start     = (const float*)d_in[3];
  const float* trans     = (const float*)d_in[4];
  const float* endt      = (const float*)d_in[5];
  float* out = (float*)d_out;

  char* ws = (char*)d_ws;
  _Float16* PT   = (_Float16*)(ws);                       // 128*128*2 = 32768 B
  float*    logZ = (float*)(ws + 32768);                  // 512*4
  float*    path = (float*)(ws + 32768 + BATCH * 4);      // 512*4

  // A: exp(transitions)^T in f16
  crf_expT_kernel<<<64, 256, 0, stream>>>(trans, PT);
  // B: forward scan with WMMA (32 WGs x 16 batch rows)
  crf_forward_kernel<<<32, 256, 0, stream>>>(emissions, mask, start, endt, PT, logZ);
  // C: gold-path scores
  crf_path_kernel<<<2, 256, 0, stream>>>(emissions, tags, mask, start, trans, endt, path);
  // D: final scalar reduction
  crf_reduce_kernel<<<1, 256, 0, stream>>>(path, logZ, out);
}
